// UncertaintyAwareQNetwork_45810121179180
// MI455X (gfx1250) — matrix operand order repaired and checked by reading, not compiled, not run on hardware
//
#include <hip/hip_runtime.h>
#include <math.h>

typedef __attribute__((ext_vector_type(2)))  float  v2f;
typedef __attribute__((ext_vector_type(8)))  float  v8f;
typedef __attribute__((ext_vector_type(4)))  __bf16 v4bf;
typedef __attribute__((ext_vector_type(8)))  __bf16 v8bf;
typedef __attribute__((ext_vector_type(16))) __bf16 v16bf;

// ---------------- problem constants ----------------
#define S_DIM 256
#define A_DIM 18
#define H_DIM 256
#define E_NUM 5
#define NS_NUM 10
#define B_NUM 4096
#define ES_NUM (E_NUM * NS_NUM)   // 50

// ---------------- workspace layout (byte offsets) ----------------
// W0h/W0l/W1h/W1l: 50*256*256 bf16 = 6,553,600 B each
// W2: 50*18*256 f32 ; B0/B1: 50*256 f32 ; B2: 50*18 f32 ; NET: 5*4096*18 f32
#define WOFF_W0H 0u
#define WOFF_W0L 6553600u
#define WOFF_W1H 13107200u
#define WOFF_W1L 19660800u
#define WOFF_W2  26214400u
#define WOFF_B0  27136000u
#define WOFF_B1  27187200u
#define WOFF_B2  27238400u
#define WOFF_NET 27242496u
#define NET_SIZE 368640u

// ---------------- output layout (float offsets) ----------------
#define OUT_QMEAN 0u
#define OUT_EPI   73728u
#define OUT_ALEA  147456u
#define OUT_TOT   221184u
#define OUT_LCB   294912u
#define OUT_UCB   368640u
#define OUT_RISK  442368u
#define OUT_KL    446464u

#define BSTR  264   // bf16 LDS row stride (conflict-free 16B frag loads)
#define FSTR  260   // f32 LDS row stride for layer-2 input
#define LDS_BYTES (64 * BSTR * 2 * 2)   // hi+lo planes = 67584 B

__device__ __forceinline__ float softplusf(float x) {
    return (x > 20.f) ? x : log1pf(expf(x));
}

__device__ __forceinline__ v16bf frag16(v8bf a, v8bf b) {
    return __builtin_shufflevector(a, b, 0,1,2,3,4,5,6,7,8,9,10,11,12,13,14,15);
}

// =====================================================================
// 0) zero the net accumulator and the KL output slot (every call)
// =====================================================================
__global__ void zero_kernel(float* __restrict__ net, float* __restrict__ out) {
    unsigned idx = blockIdx.x * blockDim.x + threadIdx.x;
    unsigned stride = gridDim.x * blockDim.x;
    for (unsigned i = idx; i <= NET_SIZE; i += stride) {
        if (i < NET_SIZE) net[i] = 0.f;
        else out[OUT_KL] = 0.f;
    }
}

// =====================================================================
// 1) sample weights:  W = mu + softplus(rho)*eps
//    layers 0/1 stored as bf16 hi/lo planes; layer2 + biases fp32
// =====================================================================
__global__ void sample_kernel(
    const float* __restrict__ wmu0, const float* __restrict__ wrho0, const float* __restrict__ weps0,
    const float* __restrict__ bmu0, const float* __restrict__ brho0, const float* __restrict__ beps0,
    const float* __restrict__ wmu1, const float* __restrict__ wrho1, const float* __restrict__ weps1,
    const float* __restrict__ bmu1, const float* __restrict__ brho1, const float* __restrict__ beps1,
    const float* __restrict__ wmu2, const float* __restrict__ wrho2, const float* __restrict__ weps2,
    const float* __restrict__ bmu2, const float* __restrict__ brho2, const float* __restrict__ beps2,
    __bf16* __restrict__ W0h, __bf16* __restrict__ W0l,
    __bf16* __restrict__ W1h, __bf16* __restrict__ W1l,
    float* __restrict__ W2,  float* __restrict__ B0,
    float* __restrict__ B1,  float* __restrict__ B2)
{
    const unsigned N0 = 3276800u, N1 = 6553600u, N2 = 6784000u;
    const unsigned NB0 = 6796800u, NB1 = 6809600u, NTOT = 6810500u;
    unsigned idx = blockIdx.x * blockDim.x + threadIdx.x;
    unsigned stride = gridDim.x * blockDim.x;
    for (unsigned i = idx; i < NTOT; i += stride) {
        if (i < N0) {                                   // W0: [E,NS,256,256]
            unsigned e = i / 655360u, r = i % 65536u, m = e * 65536u + r;
            float w = wmu0[m] + softplusf(wrho0[m]) * weps0[i];
            __bf16 h = (__bf16)w;
            W0h[i] = h; W0l[i] = (__bf16)(w - (float)h);
        } else if (i < N1) {                            // W1
            unsigned j = i - N0;
            unsigned e = j / 655360u, r = j % 65536u, m = e * 65536u + r;
            float w = wmu1[m] + softplusf(wrho1[m]) * weps1[j];
            __bf16 h = (__bf16)w;
            W1h[j] = h; W1l[j] = (__bf16)(w - (float)h);
        } else if (i < N2) {                            // W2: [E,NS,18,256] fp32
            unsigned j = i - N1;
            unsigned e = j / 46080u, r = j % 4608u, m = e * 4608u + r;
            W2[j] = wmu2[m] + softplusf(wrho2[m]) * weps2[j];
        } else if (i < NB0) {                           // b0: [E,NS,256]
            unsigned j = i - N2;
            unsigned e = j / 2560u, o = j % 256u, m = e * 256u + o;
            B0[j] = bmu0[m] + softplusf(brho0[m]) * beps0[j];
        } else if (i < NB1) {                           // b1
            unsigned j = i - NB0;
            unsigned e = j / 2560u, o = j % 256u, m = e * 256u + o;
            B1[j] = bmu1[m] + softplusf(brho1[m]) * beps1[j];
        } else {                                        // b2: [E,NS,18]
            unsigned j = i - NB1;
            unsigned e = j / 180u, o = j % 18u, m = e * 18u + o;
            B2[j] = bmu2[m] + softplusf(brho2[m]) * beps2[j];
        }
    }
}

// =====================================================================
// 2) KL divergence (sum over params, /E)
// =====================================================================
__device__ __forceinline__ float kl_term(float mu, float rho) {
    float s = softplusf(rho);
    return -logf(s) + 0.5f * (s * s + mu * mu) - 0.5f;
}

__global__ void kl_kernel(
    const float* __restrict__ wmu0, const float* __restrict__ wrho0,
    const float* __restrict__ bmu0, const float* __restrict__ brho0,
    const float* __restrict__ wmu1, const float* __restrict__ wrho1,
    const float* __restrict__ bmu1, const float* __restrict__ brho1,
    const float* __restrict__ wmu2, const float* __restrict__ wrho2,
    const float* __restrict__ bmu2, const float* __restrict__ brho2,
    float* __restrict__ out)
{
    const unsigned R0 = 327680u, R1 = 655360u, R2 = 678400u, R3 = 679680u, R4 = 680960u, RT = 681050u;
    __shared__ float red[256];
    unsigned idx = blockIdx.x * blockDim.x + threadIdx.x;
    unsigned stride = gridDim.x * blockDim.x;
    float acc = 0.f;
    for (unsigned i = idx; i < RT; i += stride) {
        float v;
        if (i < R0)      v = kl_term(wmu0[i], wrho0[i]);
        else if (i < R1) v = kl_term(wmu1[i - R0], wrho1[i - R0]);
        else if (i < R2) v = kl_term(wmu2[i - R1], wrho2[i - R1]);
        else if (i < R3) v = kl_term(bmu0[i - R2], brho0[i - R2]);
        else if (i < R4) v = kl_term(bmu1[i - R3], brho1[i - R3]);
        else             v = kl_term(bmu2[i - R4], brho2[i - R4]);
        acc += v;
    }
    red[threadIdx.x] = acc;
    __syncthreads();
    for (int s = 128; s > 0; s >>= 1) {
        if ((int)threadIdx.x < s) red[threadIdx.x] += red[threadIdx.x + s];
        __syncthreads();
    }
    if (threadIdx.x == 0) atomicAdd(&out[OUT_KL], red[0] * (1.0f / E_NUM));
}

// =====================================================================
// 3) fused 3-layer MLP. Layers 0/1: split-bf16 3-product emulation on
//    v_wmma_f32_16x16x32_bf16 (K=32/instr). Layer 2: fp32 16x16x4.
//    4 waves/block; wave owns 64 rows x 64-col slice.
// =====================================================================
__global__ __launch_bounds__(128)
void mlp_kernel(const float* __restrict__ state,
                const __bf16* __restrict__ W0h, const __bf16* __restrict__ W0l,
                const __bf16* __restrict__ W1h, const __bf16* __restrict__ W1l,
                const float* __restrict__ W2g,
                const float* __restrict__ B0g, const float* __restrict__ B1g,
                const float* __restrict__ B2g,
                float* __restrict__ netsum)
{
    extern __shared__ char smem[];
    __bf16* ldsH = (__bf16*)smem;            // [64][BSTR] hi plane
    __bf16* ldsL = ldsH + 64 * BSTR;         // [64][BSTR] lo plane
    float*  ldsF = (float*)smem;             // [64][FSTR] layer-2 input (aliases)

    const int lane = threadIdx.x & 31;
    const int wv   = threadIdx.x >> 5;
    const int es   = blockIdx.y;             // 0..49
    const int b0   = blockIdx.x * 64;

    const __bf16* Wh0 = W0h + (size_t)es * 65536u;
    const __bf16* Wl0 = W0l + (size_t)es * 65536u;
    const __bf16* Wh1 = W1h + (size_t)es * 65536u;
    const __bf16* Wl1 = W1l + (size_t)es * 65536u;
    const float*  W2  = W2g + (size_t)es * 4608u;
    const float*  Bv0 = B0g + es * 256;
    const float*  Bv1 = B1g + es * 256;
    const float*  Bv2 = B2g + es * 18;

    // ---- stage x tile [64][256], split into bf16 hi/lo planes ----
    {
        int row   = threadIdx.x >> 1;             // 0..63
        int halfc = (threadIdx.x & 1) * 128;
        const float4* src = (const float4*)(state + (size_t)(b0 + row) * 256 + halfc);
        #pragma unroll
        for (int i = 0; i < 32; i++) {
            float4 v = src[i];
            float xs[4] = {v.x, v.y, v.z, v.w};
            v4bf hi, lo;
            #pragma unroll
            for (int c = 0; c < 4; c++) {
                __bf16 h = (__bf16)xs[c];
                hi[c] = h;
                lo[c] = (__bf16)(xs[c] - (float)h);
            }
            *(v4bf*)&ldsH[row * BSTR + halfc + i * 4] = hi;
            *(v4bf*)&ldsL[row * BSTR + halfc + i * 4] = lo;
        }
    }
    __syncthreads();

    const int l15   = lane & 15;
    const int kA    = (lane >> 4) << 4;    // A-frag K split: +0 / +16
    const int kB    = (lane >> 4) << 4;    // B-frag K split: +0 / +16
    const int khalf = (lane >> 4) << 1;    // fp32 frag K split: +0 / +2
    const int mhalf = (lane >> 4) << 3;    // D-matrix row split

    v8f acc[4][4];

    // ---- layers 0 and 1: [64,256] x [256,256]^T via split-bf16 ----
    for (int layer = 0; layer < 2; ++layer) {
        const __bf16* Wh = layer ? Wh1 : Wh0;
        const __bf16* Wl = layer ? Wl1 : Wl0;
        const float*  Bv = layer ? Bv1 : Bv0;

        #pragma unroll
        for (int t = 0; t < 4; t++) {
            float bb = Bv[wv * 64 + t * 16 + l15];
            #pragma unroll
            for (int rg = 0; rg < 4; rg++)
                #pragma unroll
                for (int r = 0; r < 8; r++) acc[rg][t][r] = bb;
        }

        #pragma unroll 2
        for (int k0 = 0; k0 < 256; k0 += 32) {
            v16bf aH[4], aL[4];
            #pragma unroll
            for (int rg = 0; rg < 4; rg++) {
                const __bf16* pH = ldsH + (rg * 16 + l15) * BSTR + k0 + kA;
                const __bf16* pL = ldsL + (rg * 16 + l15) * BSTR + k0 + kA;
                aH[rg] = frag16(*(const v8bf*)pH, *(const v8bf*)(pH + 8));
                aL[rg] = frag16(*(const v8bf*)pL, *(const v8bf*)(pL + 8));
            }
            #pragma unroll
            for (int t = 0; t < 4; t++) {
                size_t woff = (size_t)(wv * 64 + t * 16 + l15) * 256 + k0 + kB;
                v16bf bH = *(const v16bf*)(Wh + woff);
                v16bf bL = *(const v16bf*)(Wl + woff);
                #pragma unroll
                for (int rg = 0; rg < 4; rg++) {
                    acc[rg][t] = __builtin_amdgcn_wmma_f32_16x16x32_bf16(
                        false, aH[rg], false, bL, (short)0, acc[rg][t], false, false);
                    acc[rg][t] = __builtin_amdgcn_wmma_f32_16x16x32_bf16(
                        false, aL[rg], false, bH, (short)0, acc[rg][t], false, false);
                    acc[rg][t] = __builtin_amdgcn_wmma_f32_16x16x32_bf16(
                        false, aH[rg], false, bH, (short)0, acc[rg][t], false, false);
                }
            }
        }

        __syncthreads();   // all waves done reading this layer's activations
        if (layer == 0) {
            // relu + re-split to bf16 hi/lo for layer 1
            #pragma unroll
            for (int rg = 0; rg < 4; rg++)
                #pragma unroll
                for (int t = 0; t < 4; t++)
                    #pragma unroll
                    for (int r = 0; r < 8; r++) {
                        float v = acc[rg][t][r];
                        v = v > 0.f ? v : 0.f;
                        int m = rg * 16 + r + mhalf;
                        int n = wv * 64 + t * 16 + l15;
                        __bf16 h = (__bf16)v;
                        ldsH[m * BSTR + n] = h;
                        ldsL[m * BSTR + n] = (__bf16)(v - (float)h);
                    }
        } else {
            // relu + fp32 for layer 2
            #pragma unroll
            for (int rg = 0; rg < 4; rg++)
                #pragma unroll
                for (int t = 0; t < 4; t++)
                    #pragma unroll
                    for (int r = 0; r < 8; r++) {
                        float v = acc[rg][t][r];
                        v = v > 0.f ? v : 0.f;
                        ldsF[(rg * 16 + r + mhalf) * FSTR + wv * 64 + t * 16 + l15] = v;
                    }
        }
        __syncthreads();
    }

    // ---- layer 2: [64,256] x [18,256]^T fp32 ; wave owns rows wv*16..+15 ----
    v8f acc2[2];
    #pragma unroll
    for (int t = 0; t < 2; t++) {
        int n = t * 16 + l15;
        float bb = (n < A_DIM) ? Bv2[n] : 0.f;
        #pragma unroll
        for (int r = 0; r < 8; r++) acc2[t][r] = bb;
    }
    #pragma unroll 2
    for (int k0 = 0; k0 < 256; k0 += 4) {
        int kk = k0 + khalf;
        v2f a = *(const v2f*)&ldsF[(wv * 16 + l15) * FSTR + kk];
        v2f bw[2];
        #pragma unroll
        for (int t = 0; t < 2; t++) {
            int n = t * 16 + l15;
            if (n < A_DIM) bw[t] = *(const v2f*)&W2[(size_t)n * 256 + kk];
            else { bw[t][0] = 0.f; bw[t][1] = 0.f; }
        }
        #pragma unroll
        for (int t = 0; t < 2; t++)
            acc2[t] = __builtin_amdgcn_wmma_f32_16x16x4_f32(
                false, a, false, bw[t], (short)0, acc2[t], false, false);
    }

    // accumulate MC-sample mean into net[e][b][a]
    const int e = es / NS_NUM;
    const float inv_ns = 1.0f / NS_NUM;
    #pragma unroll
    for (int t = 0; t < 2; t++) {
        int n = t * 16 + l15;
        if (n < A_DIM) {
            #pragma unroll
            for (int r = 0; r < 8; r++) {
                int m = wv * 16 + r + mhalf;
                atomicAdd(&netsum[((size_t)e * B_NUM + (b0 + m)) * A_DIM + n],
                          acc2[t][r] * inv_ns);
            }
        }
    }
}

// =====================================================================
// 4) finalize: ensemble stats + uncertainty/risk heads, one row per block
// =====================================================================
__global__ __launch_bounds__(256)
void finalize_kernel(const float* __restrict__ net,
                     const float* __restrict__ uh_w1, const float* __restrict__ uh_b1,
                     const float* __restrict__ uh_w2, const float* __restrict__ uh_b2,
                     const float* __restrict__ rh_w1, const float* __restrict__ rh_b1,
                     const float* __restrict__ rh_w2, const float* __restrict__ rh_b2,
                     float* __restrict__ out)
{
    __shared__ float uin[2 * A_DIM];
    __shared__ float rin[3 * A_DIM];
    __shared__ float h[H_DIM];
    __shared__ float red[256];
    const int b = blockIdx.x;
    const int tid = threadIdx.x;

    if (tid < A_DIM) {
        float v[E_NUM], s = 0.f;
        #pragma unroll
        for (int e = 0; e < E_NUM; e++) {
            v[e] = net[((size_t)e * B_NUM + b) * A_DIM + tid];
            s += v[e];
        }
        float mean = s * (1.0f / E_NUM);
        float var = 0.f;
        #pragma unroll
        for (int e = 0; e < E_NUM; e++) { float d = v[e] - mean; var += d * d; }
        var *= (1.0f / (E_NUM - 1));   // unbiased
        uin[tid] = mean; uin[A_DIM + tid] = var;
        rin[tid] = mean; rin[A_DIM + tid] = var;
        out[OUT_QMEAN + (size_t)b * A_DIM + tid] = mean;
        out[OUT_EPI   + (size_t)b * A_DIM + tid] = var;
    }
    __syncthreads();

    {   // uncertainty head hidden layer
        float a = uh_b1[tid];
        #pragma unroll
        for (int i = 0; i < 2 * A_DIM; i++) a += uin[i] * uh_w1[tid * (2 * A_DIM) + i];
        h[tid] = a > 0.f ? a : 0.f;
    }
    __syncthreads();

    if (tid < A_DIM) {
        float a = uh_b2[tid];
        for (int j = 0; j < H_DIM; j++) a += h[j] * uh_w2[tid * H_DIM + j];
        float alea = softplusf(a);
        rin[2 * A_DIM + tid] = alea;
        float tot = uin[A_DIM + tid] + alea;
        float sp2 = 2.0f * sqrtf(tot);
        out[OUT_ALEA + (size_t)b * A_DIM + tid] = alea;
        out[OUT_TOT  + (size_t)b * A_DIM + tid] = tot;
        out[OUT_LCB  + (size_t)b * A_DIM + tid] = uin[tid] - sp2;
        out[OUT_UCB  + (size_t)b * A_DIM + tid] = uin[tid] + sp2;
    }
    __syncthreads();

    {   // risk head hidden layer + dot with rh_w2
        float c = rh_b1[tid];
        #pragma unroll
        for (int i = 0; i < 3 * A_DIM; i++) c += rin[i] * rh_w1[tid * (3 * A_DIM) + i];
        float hv = c > 0.f ? c : 0.f;
        red[tid] = hv * rh_w2[tid];
    }
    __syncthreads();
    for (int s = 128; s > 0; s >>= 1) {
        if (tid < s) red[tid] += red[tid + s];
        __syncthreads();
    }
    if (tid == 0) out[OUT_RISK + b] = red[0] + rh_b2[0];
}

// =====================================================================
extern "C" void kernel_launch(void* const* d_in, const int* in_sizes, int n_in,
                              void* d_out, int out_size, void* d_ws, size_t ws_size,
                              hipStream_t stream)
{
    (void)in_sizes; (void)n_in; (void)out_size; (void)ws_size;
    const float* state  = (const float*)d_in[0];
    const float* w_mu0  = (const float*)d_in[2];
    const float* w_rho0 = (const float*)d_in[3];
    const float* b_mu0  = (const float*)d_in[4];
    const float* b_rho0 = (const float*)d_in[5];
    const float* w_eps0 = (const float*)d_in[6];
    const float* b_eps0 = (const float*)d_in[7];
    const float* w_mu1  = (const float*)d_in[8];
    const float* w_rho1 = (const float*)d_in[9];
    const float* b_mu1  = (const float*)d_in[10];
    const float* b_rho1 = (const float*)d_in[11];
    const float* w_eps1 = (const float*)d_in[12];
    const float* b_eps1 = (const float*)d_in[13];
    const float* w_mu2  = (const float*)d_in[14];
    const float* w_rho2 = (const float*)d_in[15];
    const float* b_mu2  = (const float*)d_in[16];
    const float* b_rho2 = (const float*)d_in[17];
    const float* w_eps2 = (const float*)d_in[18];
    const float* b_eps2 = (const float*)d_in[19];
    const float* uh_w1  = (const float*)d_in[20];
    const float* uh_b1  = (const float*)d_in[21];
    const float* uh_w2  = (const float*)d_in[22];
    const float* uh_b2  = (const float*)d_in[23];
    const float* rh_w1  = (const float*)d_in[24];
    const float* rh_b1  = (const float*)d_in[25];
    const float* rh_w2  = (const float*)d_in[26];
    const float* rh_b2  = (const float*)d_in[27];

    float* out = (float*)d_out;
    char*  wsb = (char*)d_ws;

    __bf16* W0h = (__bf16*)(wsb + WOFF_W0H);
    __bf16* W0l = (__bf16*)(wsb + WOFF_W0L);
    __bf16* W1h = (__bf16*)(wsb + WOFF_W1H);
    __bf16* W1l = (__bf16*)(wsb + WOFF_W1L);
    float*  W2  = (float*)(wsb + WOFF_W2);
    float*  B0  = (float*)(wsb + WOFF_B0);
    float*  B1  = (float*)(wsb + WOFF_B1);
    float*  B2  = (float*)(wsb + WOFF_B2);
    float*  NET = (float*)(wsb + WOFF_NET);

    (void)hipFuncSetAttribute((const void*)mlp_kernel,
                              hipFuncAttributeMaxDynamicSharedMemorySize, LDS_BYTES);

    zero_kernel<<<(NET_SIZE + 256) / 256, 256, 0, stream>>>(NET, out);

    sample_kernel<<<2048, 256, 0, stream>>>(
        w_mu0, w_rho0, w_eps0, b_mu0, b_rho0, b_eps0,
        w_mu1, w_rho1, w_eps1, b_mu1, b_rho1, b_eps1,
        w_mu2, w_rho2, w_eps2, b_mu2, b_rho2, b_eps2,
        W0h, W0l, W1h, W1l, W2, B0, B1, B2);

    kl_kernel<<<512, 256, 0, stream>>>(
        w_mu0, w_rho0, b_mu0, b_rho0,
        w_mu1, w_rho1, b_mu1, b_rho1,
        w_mu2, w_rho2, b_mu2, b_rho2, out);

    mlp_kernel<<<dim3(B_NUM / 64, ES_NUM), 128, LDS_BYTES, stream>>>(
        state, W0h, W0l, W1h, W1l, W2, B0, B1, B2, NET);

    finalize_kernel<<<B_NUM, 256, 0, stream>>>(
        NET, uh_w1, uh_b1, uh_w2, uh_b2,
        rh_w1, rh_b1, rh_w2, rh_b2, out);
}
